// HierarchicalLogits_13554916786432
// MI455X (gfx1250) — compile-verified, matrix operand order checked by reading, not run on hardware
//
#include <hip/hip_runtime.h>
#include <hip/hip_fp16.h>

typedef __attribute__((ext_vector_type(16))) _Float16 v16h;
typedef __attribute__((ext_vector_type(8)))  _Float16 v8h;
typedef __attribute__((ext_vector_type(8)))  float    v8f;

#define TOKENS 8192
#define DDIM   2048
#define NOUT   4096

#define BM 128
#define BN 128
#define BK 32
#define LDS_STRIDE 40   // halfs per row (32 data + 8 pad), rows stay 16B aligned

// ---------------------------------------------------------------------------
// Kernel 1: convert X (fp32) -> fp16 into workspace
// ---------------------------------------------------------------------------
__global__ __launch_bounds__(256)
void cvt_x_kernel(const float* __restrict__ X, _Float16* __restrict__ Xh) {
    const int total = (TOKENS * DDIM) / 4;
    for (int i = blockIdx.x * blockDim.x + threadIdx.x; i < total;
         i += gridDim.x * blockDim.x) {
        const float4 v = ((const float4*)X)[i];
        union { _Float16 h[4]; uint2 u; } p;
        p.h[0] = (_Float16)v.x; p.h[1] = (_Float16)v.y;
        p.h[2] = (_Float16)v.z; p.h[3] = (_Float16)v.w;
        ((uint2*)Xh)[i] = p.u;
    }
}

// ---------------------------------------------------------------------------
// Kernel 2: build W^T (NOUT x DDIM) in fp16.
// W[:,n] = sum over the 12 preorder ancestors r of leaf n of sign * normals[r,:]
// Ancestor indices/signs derived from the perfect-binary-tree preorder layout:
// right child of node r (subtree size s) is at r + s/2; left child at r + 1.
// ---------------------------------------------------------------------------
__global__ __launch_bounds__(256)
void build_w_kernel(const float* __restrict__ normals, _Float16* __restrict__ WTh) {
    const int n = blockIdx.x;            // leaf index, uniform per block (SALU path)
    int rows[12];
    float signs[12];
    int lo = 0, size = NOUT, r = 0;
#pragma unroll
    for (int k = 0; k < 12; ++k) {
        const int half = size >> 1;
        rows[k] = r;
        if (n < lo + half) { signs[k] =  1.0f; r = r + 1; }
        else               { signs[k] = -1.0f; r = r + half; lo += half; }
        size = half;
    }
    for (int d = threadIdx.x; d < DDIM; d += blockDim.x) {
        float acc = 0.0f;
#pragma unroll
        for (int k = 0; k < 12; ++k)
            acc += signs[k] * normals[(size_t)rows[k] * DDIM + d];
        WTh[(size_t)n * DDIM + d] = (_Float16)acc;
    }
}

// ---------------------------------------------------------------------------
// Fragment loader: K-major LDS tile -> WMMA A/B fragment (16x32 f16).
// Lane L: row = base + L%16 ; chunks at K = 8*(L/16) and 16 + 8*(L/16).
// ---------------------------------------------------------------------------
__device__ __forceinline__ v16h frag_load(const _Float16* tile, int row, int group) {
    const _Float16* p = tile + row * LDS_STRIDE;
    const v8h lo = *(const v8h*)(p + group * 8);
    const v8h hi = *(const v8h*)(p + 16 + group * 8);
    v16h f;
#pragma unroll
    for (int e = 0; e < 8; ++e) { f[e] = lo[e]; f[e + 8] = hi[e]; }
    return f;
}

// ---------------------------------------------------------------------------
// Kernel 3: out[t][n] = sum_d Xh[t][d] * WTh[n][d] + bias[n]   (WMMA f16->f32)
// ---------------------------------------------------------------------------
__global__ __launch_bounds__(256)
void gemm_wmma_kernel(const _Float16* __restrict__ A,   // Xh  [TOKENS][DDIM]
                      const _Float16* __restrict__ B,   // WTh [NOUT][DDIM]
                      const float* __restrict__ bias,
                      float* __restrict__ out) {
    __shared__ _Float16 sA[BM * LDS_STRIDE];
    __shared__ _Float16 sB[BN * LDS_STRIDE];

    const int tid   = threadIdx.x;
    const int wave  = tid >> 5;
    const int lane  = tid & 31;
    const int lrow  = lane & 15;
    const int group = lane >> 4;

    const int bm = blockIdx.x * BM;
    const int bn = blockIdx.y * BN;

    const int wm = (wave & 3) * 32;   // wave's M offset: 2 frags of 16
    const int wn = (wave >> 2) * 64;  // wave's N offset: 4 frags of 16

    v8f acc[2][4];
#pragma unroll
    for (int i = 0; i < 2; ++i)
#pragma unroll
        for (int j = 0; j < 4; ++j)
#pragma unroll
            for (int e = 0; e < 8; ++e) acc[i][j][e] = 0.0f;

    const int ldr = tid >> 2;        // 0..63  (row within a 64-row pass)
    const int ldc = (tid & 3) * 8;   // 0,8,16,24 halfs (16B chunks)

    for (int kt = 0; kt < DDIM; kt += BK) {
#pragma unroll
        for (int p = 0; p < 2; ++p) {
            const int r = ldr + p * 64;
            const uint4 va = *(const uint4*)(A + (size_t)(bm + r) * DDIM + kt + ldc);
            *(uint4*)(sA + r * LDS_STRIDE + ldc) = va;
            const uint4 vb = *(const uint4*)(B + (size_t)(bn + r) * DDIM + kt + ldc);
            *(uint4*)(sB + r * LDS_STRIDE + ldc) = vb;
        }
        __syncthreads();

        v16h afrag[2], bfrag[4];
#pragma unroll
        for (int i = 0; i < 2; ++i)
            afrag[i] = frag_load(sA, wm + i * 16 + lrow, group);
#pragma unroll
        for (int j = 0; j < 4; ++j)
            bfrag[j] = frag_load(sB, wn + j * 16 + lrow, group);

#pragma unroll
        for (int i = 0; i < 2; ++i)
#pragma unroll
            for (int j = 0; j < 4; ++j)
                acc[i][j] = __builtin_amdgcn_wmma_f32_16x16x32_f16(
                    /*neg_a=*/false, afrag[i],
                    /*neg_b=*/false, bfrag[j],
                    /*c_mod=*/(short)0, acc[i][j],
                    /*reuse_a=*/false, /*reuse_b=*/false);
        __syncthreads();
    }

    // C/D layout: VGPR jj -> row jj + 8*group ; col = lane%16
#pragma unroll
    for (int i = 0; i < 2; ++i) {
#pragma unroll
        for (int jf = 0; jf < 4; ++jf) {
            const int col = bn + wn + jf * 16 + lrow;
            const float bv = bias[col];
            const int rbase = bm + wm + i * 16 + 8 * group;
#pragma unroll
            for (int jj = 0; jj < 8; ++jj)
                out[(size_t)(rbase + jj) * NOUT + col] = acc[i][jf][jj] + bv;
        }
    }
}

// ---------------------------------------------------------------------------
extern "C" void kernel_launch(void* const* d_in, const int* in_sizes, int n_in,
                              void* d_out, int out_size, void* d_ws, size_t ws_size,
                              hipStream_t stream) {
    const float* X       = (const float*)d_in[0];  // (8192, 2048)
    const float* normals = (const float*)d_in[1];  // (4095, 2048)
    const float* biases  = (const float*)d_in[2];  // (4096,)
    // d_in[3] = S : structure is deterministic (perfect tree), recomputed analytically.
    float* out = (float*)d_out;                    // (8192, 4096)

    _Float16* Xh  = (_Float16*)d_ws;                       // 32 MB
    _Float16* WTh = Xh + (size_t)TOKENS * DDIM;            // +16 MB

    cvt_x_kernel<<<4096, 256, 0, stream>>>(X, Xh);
    build_w_kernel<<<NOUT, 256, 0, stream>>>(normals, WTh);

    dim3 grid(TOKENS / BM, NOUT / BN);   // 64 x 32
    gemm_wmma_kernel<<<grid, 256, 0, stream>>>(Xh, WTh, biases, out);
}
